// BaseAttention_62019327754726
// MI455X (gfx1250) — matrix operand order repaired and checked by reading, not compiled, not run on hardware
//
#include <hip/hip_runtime.h>

#define B_ 4
#define H_ 16
#define T_ 2048
#define D_ 128
#define W_ 512
// SCALE * log2(e): softmax done in exp2 domain (saves a v_mul per element)
#define QSCALE_ 0.1275197644252955f   // 128^-0.5 * 1.4426950408889634

#define SK_STRIDE 136                  // bf16 elems per K row (272B: 16B aligned, bank step 4)
#define SV_STRIDE 72                   // bf16 elems per Vt row (144B: 16B aligned, bank step 36)
#define SK_ELEMS  (32 * SK_STRIDE)
#define SV_ELEMS  (128 * SV_STRIDE)

typedef __attribute__((ext_vector_type(16))) __bf16 v16bf;
typedef __attribute__((ext_vector_type(8)))  __bf16 v8bf;
typedef __attribute__((ext_vector_type(4)))  __bf16 v4bf;
typedef __attribute__((ext_vector_type(8)))  float  v8f;
typedef __attribute__((ext_vector_type(4)))  float  v4f;

static __device__ __forceinline__ __bf16 tobf(float x) { return (__bf16)x; }

struct StageRegs {
    v4f   k[8];    // 32 K floats per thread (row-major tile)
    float v[32];   // 32 V floats per thread (one d-column, 32 keys)
};

static __device__ __forceinline__ void load_regs(const float* __restrict__ Kp,
                                                 const float* __restrict__ Vp,
                                                 int c, int tid, StageRegs& r) {
#pragma unroll
    for (int i = 0; i < 8; ++i) {
        const int idx = (i * 128 + tid) << 2;   // 0..4095 step 4
        const int row = idx >> 7;               // key 0..31
        const int col = idx & 127;              // d 0..127
        r.k[i] = *(const v4f*)(Kp + (size_t)(c + row) * D_ + col);
    }
#pragma unroll
    for (int i = 0; i < 32; ++i)                // lanes read consecutive d: coalesced
        r.v[i] = Vp[(size_t)(c + i) * D_ + tid];
}

static __device__ __forceinline__ void store_lds(__bf16* __restrict__ sKc,
                                                 __bf16* __restrict__ sVc,
                                                 int tid, const StageRegs& r) {
#pragma unroll
    for (int i = 0; i < 8; ++i) {
        const int idx = (i * 128 + tid) << 2;
        const int row = idx >> 7;
        const int col = idx & 127;
        v4bf p;
#pragma unroll
        for (int j = 0; j < 4; ++j) p[j] = tobf(r.k[i][j]);
        *(v4bf*)(sKc + row * SK_STRIDE + col) = p;   // 8B write, banks 2t,2t+1: conflict-free
    }
#pragma unroll
    for (int rb = 0; rb < 8; ++rb) {                 // Vt[d=tid][key]: 8B packed writes
        v4bf p;
#pragma unroll
        for (int j = 0; j < 4; ++j) p[j] = tobf(r.v[rb * 4 + j]);
        *(v4bf*)(sVc + tid * SV_STRIDE + rb * 4) = p;
    }
}

__global__ __launch_bounds__(128) void swa_fa_kernel(
    const float* __restrict__ Q, const float* __restrict__ K,
    const float* __restrict__ V, float* __restrict__ O)
{
    __shared__ __bf16 sK[2 * SK_ELEMS];   // double-buffered K chunk, row-major [key][d]
    __shared__ __bf16 sV[2 * SV_ELEMS];   // double-buffered V chunk, transposed [d][key]

    const int tid  = threadIdx.x;
    const int lane = tid & 31;
    const int wave = tid >> 5;
    const int hi   = (lane >> 4) & 1;
    const int ln   = lane & 15;

    const int q0 = blockIdx.x * 64;
    const int bh = blockIdx.y;
    const size_t base = (size_t)bh * T_ * D_;
    const float* Qp = Q + base;
    const float* Kp = K + base;
    const float* Vp = V + base;
    float*       Op = O + base;

    const int qw   = q0 + wave * 16;
    const int qrow = qw + ln;              // this lane's q (column of S^T / O^T)

    // ---- Q^T fragments (B operand of S^T = K * Q^T), scale*log2e folded in ----
    v16bf qf[4];
    {
        const float* qr = Qp + (size_t)qrow * D_;
#pragma unroll
        for (int dc = 0; dc < 4; ++dc) {
            const v4f* p4 = (const v4f*)(qr + dc * 32 + hi * 16);
#pragma unroll
            for (int v = 0; v < 4; ++v) {
                v4f f = p4[v];
#pragma unroll
                for (int e = 0; e < 4; ++e) qf[dc][v * 4 + e] = tobf(f[e] * QSCALE_);
            }
        }
    }

    v8f o[8];
#pragma unroll
    for (int t = 0; t < 8; ++t)
#pragma unroll
        for (int r = 0; r < 8; ++r) o[t][r] = 0.0f;

    float mrow = -1e30f, lrow = 0.0f;

    int kbeg = q0 - (W_ - 1);
    if (kbeg < 0) kbeg = 0;
    kbeg &= ~31;
    const int kend = q0 + 64;

    // ---- software pipeline: stage chunk 0 ----
    StageRegs rg;
    load_regs(Kp, Vp, kbeg, tid, rg);
    store_lds(sK, sV, tid, rg);

    int ibuf = 0;
    for (int c = kbeg; c < kend; c += 32) {
        __syncthreads();                       // buffer ibuf visible to all waves
        const bool more = (c + 32 < kend);
        if (more) load_regs(Kp, Vp, c + 32, tid, rg);   // issue next-chunk global loads early
        if (c + 64 < kend) {                             // L2 prefetch two chunks ahead
            __builtin_prefetch(Kp + (size_t)(c + 64 + (tid & 31)) * D_ + (tid >> 5) * 32, 0, 3);
            __builtin_prefetch(Vp + (size_t)(c + 64 + (tid & 31)) * D_ + (tid >> 5) * 32, 0, 3);
        }

        const __bf16* sKc = sK + ibuf * SK_ELEMS;
        const __bf16* sVc = sV + ibuf * SV_ELEMS;

        // wave-uniform: does this chunk intersect this wave's mask at all?
        const bool anyvalid = (c <= qw + 15) && (c + 31 >= qw - (W_ - 1));
        if (anyvalid) {
            // ---- S^T = K * Q^T : two 16x16 tiles ----
            v8f st0, st1;
#pragma unroll
            for (int r = 0; r < 8; ++r) { st0[r] = 0.0f; st1[r] = 0.0f; }
#pragma unroll
            for (int dc = 0; dc < 4; ++dc) {
                const __bf16* kr0 = sKc + ln * SK_STRIDE        + dc * 32 + hi * 16;
                const __bf16* kr1 = sKc + (16 + ln) * SK_STRIDE + dc * 32 + hi * 16;
                v8bf a0l = *(const v8bf*)kr0, a0h = *(const v8bf*)(kr0 + 8);
                v8bf a1l = *(const v8bf*)kr1, a1h = *(const v8bf*)(kr1 + 8);
                v16bf a0, a1;
#pragma unroll
                for (int e = 0; e < 8; ++e) {
                    a0[e] = a0l[e]; a0[8 + e] = a0h[e];
                    a1[e] = a1l[e]; a1[8 + e] = a1h[e];
                }
                st0 = __builtin_amdgcn_wmma_f32_16x16x32_bf16(false, a0, false, qf[dc],
                                                              (short)0, st0, false, false);
                st1 = __builtin_amdgcn_wmma_f32_16x16x32_bf16(false, a1, false, qf[dc],
                                                              (short)0, st1, false, false);
            }

            // wave-uniform: interior chunk (no per-element masking needed)?
            const bool fullvalid = (c + 31 <= qw) && (c >= qw + 15 - (W_ - 1));

            float p0[8], p1[8];
            float mloc = -1e30f;
            if (fullvalid) {
#pragma unroll
                for (int r = 0; r < 8; ++r) {
                    p0[r] = st0[r]; p1[r] = st1[r];
                    mloc = fmaxf(mloc, fmaxf(p0[r], p1[r]));
                }
            } else {
#pragma unroll
                for (int r = 0; r < 8; ++r) {
                    const int k0 = c + r + hi * 8;
                    const int k1 = k0 + 16;
                    const bool v0 = (k0 <= qrow) && (qrow - k0 < W_);
                    const bool v1 = (k1 <= qrow) && (qrow - k1 < W_);
                    p0[r] = v0 ? st0[r] : -1e30f;
                    p1[r] = v1 ? st1[r] : -1e30f;
                    mloc = fmaxf(mloc, fmaxf(p0[r], p1[r]));
                }
            }
            mloc = fmaxf(mloc, __shfl_xor(mloc, 16, 32));
            const float mnew  = fmaxf(mrow, mloc);
            const float alpha = exp2f(mrow - mnew);
            float lloc = 0.0f;
            if (fullvalid) {
#pragma unroll
                for (int r = 0; r < 8; ++r) {
                    p0[r] = exp2f(p0[r] - mnew);
                    p1[r] = exp2f(p1[r] - mnew);
                    lloc += p0[r] + p1[r];
                }
            } else {
#pragma unroll
                for (int r = 0; r < 8; ++r) {
                    p0[r] = (p0[r] > -1e29f) ? exp2f(p0[r] - mnew) : 0.0f;
                    p1[r] = (p1[r] > -1e29f) ? exp2f(p1[r] - mnew) : 0.0f;
                    lloc += p0[r] + p1[r];
                }
            }
            lloc += __shfl_xor(lloc, 16, 32);
            lrow  = lrow * alpha + lloc;
            mrow  = mnew;

#pragma unroll
            for (int t = 0; t < 8; ++t)
#pragma unroll
                for (int r = 0; r < 8; ++r) o[t][r] *= alpha;

            // ---- repack P^T (D layout) -> B operand via xor-16 half swap ----
            v16bf pb;
#pragma unroll
            for (int r = 0; r < 8; ++r) {
                const float e0 = __shfl_xor(p0[r], 16, 32);
                const float e1 = __shfl_xor(p1[r], 16, 32);
                pb[r]     = tobf(hi ? e1    : p0[r]);
                pb[8 + r] = tobf(hi ? p1[r] : e0);
            }

            // ---- O^T += V^T * P^T ----
#pragma unroll
            for (int t = 0; t < 8; ++t) {
                const __bf16* vr = sVc + (t * 16 + ln) * SV_STRIDE + hi * 16;
                v8bf vl = *(const v8bf*)vr, vh = *(const v8bf*)(vr + 8);
                v16bf av;
#pragma unroll
                for (int e = 0; e < 8; ++e) { av[e] = vl[e]; av[8 + e] = vh[e]; }
                o[t] = __builtin_amdgcn_wmma_f32_16x16x32_bf16(false, av, false, pb,
                                                               (short)0, o[t], false, false);
            }
        }

        if (more) store_lds(sK + (ibuf ^ 1) * SK_ELEMS, sV + (ibuf ^ 1) * SV_ELEMS, tid, rg);
        ibuf ^= 1;
    }

    // ---- normalize and store ----
    const float inv = 1.0f / lrow;
#pragma unroll
    for (int t = 0; t < 8; ++t) {
        float* p = Op + (size_t)qrow * D_ + t * 16 + hi * 8;
        v4f lo4, hi4;
#pragma unroll
        for (int r = 0; r < 4; ++r) { lo4[r] = o[t][r] * inv; hi4[r] = o[t][4 + r] * inv; }
        *(v4f*)(p)     = lo4;
        *(v4f*)(p + 4) = hi4;
    }
}

extern "C" void kernel_launch(void* const* d_in, const int* in_sizes, int n_in,
                              void* d_out, int out_size, void* d_ws, size_t ws_size,
                              hipStream_t stream) {
    const float* q = (const float*)d_in[0];
    const float* k = (const float*)d_in[1];
    const float* v = (const float*)d_in[2];
    float* out = (float*)d_out;
    dim3 grid(T_ / 64, B_ * H_);
    swa_fa_kernel<<<grid, 128, 0, stream>>>(q, k, v, out);
}
